// SiglipAttention_27650999451938
// MI455X (gfx1250) — 16-bit and 8-bit matrix operand order repaired and checked by reading, compiled, not run on hardware
//
#include <hip/hip_runtime.h>
#include <hip/hip_bf16.h>
#include <stdint.h>

#define B_  16
#define S_  729
#define H_  1152
#define NH_ 16
#define HD_ 72
#define BS_ (B_*S_)
#define SP_ 736
#define KP_ 128
#define DP_ 80
#define QT_ 46
#define NTILE_ (H_/16)
#define MTILE_ (BS_/16)
#define NGRP_ (NTILE_/4)
#define ATT_SCALE 0.11785113019775793f

#define SL_X 0
#define SL_WQ 1
#define SL_WK 2
#define SL_WV 3
#define SL_WO 4
#define SL_Q 5
#define SL_K 6
#define SL_LOG 7
#define SL_ZMIN 8
#define SL_O 9

typedef int8_t i8;
typedef int      v8i  __attribute__((ext_vector_type(8)));
typedef float    v8f  __attribute__((ext_vector_type(8)));
typedef _Float16 v16h __attribute__((ext_vector_type(16)));

__device__ __forceinline__ float scale_from(const unsigned* Sc, int slot, float qmax) {
  float a = __uint_as_float(Sc[slot]);
  return fmaxf(a, 1e-8f) / qmax;
}

__device__ __forceinline__ v8i load_a_i8(const i8* __restrict__ base, int lda,
                                         int row0, int k0, int lane) {
  int half = lane >> 4, m = lane & 15;
  const i8* p = base + (size_t)(row0 + m) * lda + k0 + half * 16;
  v8i a; int2 t;
  t = *(const int2*)(p +  0); a[0] = t.x; a[1] = t.y;
  t = *(const int2*)(p +  8); a[2] = t.x; a[3] = t.y;
  t = *(const int2*)(p + 32); a[4] = t.x; a[5] = t.y;
  t = *(const int2*)(p + 40); a[6] = t.x; a[7] = t.y;
  return a;
}
__device__ __forceinline__ v8i load_b_i8(const i8* __restrict__ base, int ldb,
                                         int col0, int k0, int lane) {
  int half = lane >> 4, n = lane & 15;
  const i8* p = base + (size_t)(col0 + n) * ldb + k0 + half * 16;
  int4 t0 = *(const int4*)(p + 0);
  int4 t1 = *(const int4*)(p + 32);
  v8i b;
  b[0]=t0.x; b[1]=t0.y; b[2]=t0.z; b[3]=t0.w;
  b[4]=t1.x; b[5]=t1.y; b[6]=t1.z; b[7]=t1.w;
  return b;
}

__device__ __forceinline__ float wave_max32(float v) {
  #pragma unroll
  for (int o = 16; o; o >>= 1) v = fmaxf(v, __shfl_xor(v, o, 32));
  return v;
}
__device__ __forceinline__ float wave_min32(float v) {
  #pragma unroll
  for (int o = 16; o; o >>= 1) v = fminf(v, __shfl_xor(v, o, 32));
  return v;
}
__device__ __forceinline__ float half_max16(float v) {
  #pragma unroll
  for (int o = 1; o < 16; o <<= 1) v = fmaxf(v, __shfl_xor(v, o, 16));
  return v;
}
__device__ __forceinline__ float half_sum16(float v) {
  #pragma unroll
  for (int o = 1; o < 16; o <<= 1) v += __shfl_xor(v, o, 16);
  return v;
}

__global__ void init_scales_kernel(unsigned* Sc) {
  int i = threadIdx.x;
  if (i < 16) Sc[i] = (i == SL_ZMIN) ? 0x7F800000u : 0u;
}

__global__ void absmax_kernel(const float* __restrict__ x, size_t n,
                              unsigned* Sc, int slot) {
  __shared__ float red[256];
  float m = 0.f;
  for (size_t i = (size_t)blockIdx.x * blockDim.x + threadIdx.x; i < n;
       i += (size_t)gridDim.x * blockDim.x)
    m = fmaxf(m, fabsf(x[i]));
  red[threadIdx.x] = m; __syncthreads();
  for (int s = 128; s; s >>= 1) {
    if (threadIdx.x < s) red[threadIdx.x] = fmaxf(red[threadIdx.x], red[threadIdx.x + s]);
    __syncthreads();
  }
  if (threadIdx.x == 0) atomicMax(&Sc[slot], __float_as_uint(red[0]));
}

__global__ void quant8_kernel(const float* __restrict__ x, size_t n,
                              const unsigned* __restrict__ Sc, int slot,
                              i8* __restrict__ q) {
  size_t i = (size_t)blockIdx.x * 256 + threadIdx.x;
  if (i >= n) return;
  float s = scale_from(Sc, slot, 127.f);
  float v = rintf(x[i] / s);
  v = fminf(fmaxf(v, -128.f), 127.f);
  q[i] = (i8)(int)v;
}

__global__ void quant_head_kernel(const float* __restrict__ Xf,
                                  const unsigned* __restrict__ Sc, int slot,
                                  i8* __restrict__ q) {
  size_t i = (size_t)blockIdx.x * 256 + threadIdx.x;
  int d  = (int)(i & (KP_ - 1));
  int s  = (int)((i >> 7) % SP_);
  int bh = (int)(i / ((size_t)SP_ * KP_));
  float v = 0.f;
  if (s < S_ && d < HD_)
    v = Xf[((size_t)(bh >> 4) * S_ + s) * H_ + (bh & 15) * HD_ + d];
  float sc = scale_from(Sc, slot, 127.f);
  float t = rintf(v / sc);
  t = fminf(fmaxf(t, -128.f), 127.f);
  q[i] = (i8)(int)t;
}

__global__ void gemm_i8_kernel(const i8* __restrict__ Aq, const i8* __restrict__ Bq,
                               const float* __restrict__ bias,
                               const unsigned* __restrict__ Sc, int slotA, int slotB,
                               int K, float* __restrict__ outF,
                               _Float16* __restrict__ vh, int mode) {
  int lane = threadIdx.x & 31;
  int t = blockIdx.x * 8 + (threadIdx.x >> 5);
  if (t >= MTILE_ * NGRP_) return;
  int row0 = (t / NGRP_) * 16, colg = (t % NGRP_) * 64;
  float sAB = scale_from(Sc, slotA, 127.f) * scale_from(Sc, slotB, 127.f);
  v8i acc0 = {0,0,0,0,0,0,0,0}, acc1 = acc0, acc2 = acc0, acc3 = acc0;
  for (int k0 = 0; k0 < K; k0 += 64) {
    v8i a = load_a_i8(Aq, K, row0, k0, lane);
    if (k0 + 512 < K)
      __builtin_prefetch(Aq + (size_t)(row0 + (lane & 15)) * K + k0 + 512, 0, 1);
    v8i b0 = load_b_i8(Bq, K, colg +  0, k0, lane);
    v8i b1 = load_b_i8(Bq, K, colg + 16, k0, lane);
    v8i b2 = load_b_i8(Bq, K, colg + 32, k0, lane);
    v8i b3 = load_b_i8(Bq, K, colg + 48, k0, lane);
    acc0 = __builtin_amdgcn_wmma_i32_16x16x64_iu8(true, a, true, b0, acc0, false, false);
    acc1 = __builtin_amdgcn_wmma_i32_16x16x64_iu8(true, a, true, b1, acc1, false, false);
    acc2 = __builtin_amdgcn_wmma_i32_16x16x64_iu8(true, a, true, b2, acc2, false, false);
    acc3 = __builtin_amdgcn_wmma_i32_16x16x64_iu8(true, a, true, b3, acc3, false, false);
  }
  int half = lane >> 4, m = lane & 15;
  v8i accs[4] = {acc0, acc1, acc2, acc3};
  #pragma unroll
  for (int c = 0; c < 4; ++c) {
    #pragma unroll
    for (int r = 0; r < 8; ++r) {
      int row = row0 + r + half * 8, col = colg + c * 16 + m;
      float v = (float)accs[c][r] * sAB + bias[col];
      if (mode == 0) {
        outF[(size_t)row * H_ + col] = v;
      } else {
        int b_ = row / S_, s_ = row - b_ * S_;
        int h_ = col / HD_, d_ = col - h_ * HD_;
        vh[((size_t)(b_ * NH_ + h_) * SP_ + s_) * DP_ + d_] = (_Float16)v;
      }
    }
  }
}

__global__ void attn_logit_amax_kernel(const i8* __restrict__ q8,
                                       const i8* __restrict__ k8,
                                       unsigned* Sc) {
  int lane = threadIdx.x & 31;
  int t = blockIdx.x * 8 + (threadIdx.x >> 5);
  int qt = t % QT_; int bh = t / QT_;
  const i8* qb = q8 + (size_t)bh * SP_ * KP_;
  const i8* kb = k8 + (size_t)bh * SP_ * KP_;
  float sqk = scale_from(Sc, SL_Q, 127.f) * scale_from(Sc, SL_K, 127.f);
  v8i a0 = load_a_i8(qb, KP_, qt * 16, 0, lane);
  v8i a1 = load_a_i8(qb, KP_, qt * 16, 64, lane);
  float mx = 0.f;
  for (int kt = 0; kt < QT_; ++kt) {
    v8i b0 = load_b_i8(kb, KP_, kt * 16, 0, lane);
    v8i b1 = load_b_i8(kb, KP_, kt * 16, 64, lane);
    v8i acc = {0,0,0,0,0,0,0,0};
    acc = __builtin_amdgcn_wmma_i32_16x16x64_iu8(true, a0, true, b0, acc, false, false);
    acc = __builtin_amdgcn_wmma_i32_16x16x64_iu8(true, a1, true, b1, acc, false, false);
    #pragma unroll
    for (int r = 0; r < 8; ++r) mx = fmaxf(mx, fabsf((float)acc[r] * sqk));
  }
  mx = wave_max32(mx);
  if (lane == 0) atomicMax(&Sc[SL_LOG], __float_as_uint(mx));
}

__global__ void attn_rowstats_kernel(const i8* __restrict__ q8,
                                     const i8* __restrict__ k8,
                                     unsigned* Sc,
                                     float* __restrict__ mrow,
                                     float* __restrict__ zrow) {
  int lane = threadIdx.x & 31;
  int t = blockIdx.x * 8 + (threadIdx.x >> 5);
  int qt = t % QT_; int bh = t / QT_;
  int half = lane >> 4, m = lane & 15;
  const i8* qb = q8 + (size_t)bh * SP_ * KP_;
  const i8* kb = k8 + (size_t)bh * SP_ * KP_;
  float sqk = scale_from(Sc, SL_Q, 127.f) * scale_from(Sc, SL_K, 127.f);
  float s16 = scale_from(Sc, SL_LOG, 32767.f);
  v8i a0 = load_a_i8(qb, KP_, qt * 16, 0, lane);
  v8i a1 = load_a_i8(qb, KP_, qt * 16, 64, lane);
  float m8[8], z8[8];
  #pragma unroll
  for (int r = 0; r < 8; ++r) { m8[r] = -__builtin_inff(); z8[r] = 0.f; }
  for (int kt = 0; kt < QT_; ++kt) {
    v8i b0 = load_b_i8(kb, KP_, kt * 16, 0, lane);
    v8i b1 = load_b_i8(kb, KP_, kt * 16, 64, lane);
    v8i acc = {0,0,0,0,0,0,0,0};
    acc = __builtin_amdgcn_wmma_i32_16x16x64_iu8(true, a0, true, b0, acc, false, false);
    acc = __builtin_amdgcn_wmma_i32_16x16x64_iu8(true, a1, true, b1, acc, false, false);
    int col = kt * 16 + m;
    #pragma unroll
    for (int r = 0; r < 8; ++r) {
      float lq = rintf(((float)acc[r] * sqk) / s16);
      lq = fminf(fmaxf(lq, -32768.f), 32767.f);
      float l = lq * s16 * ATT_SCALE;
      float lv = (col < S_) ? l : -__builtin_inff();
      float mn = fmaxf(m8[r], half_max16(lv));
      float e = (col < S_) ? __expf(lv - mn) : 0.f;
      z8[r] = z8[r] * __expf(m8[r] - mn) + half_sum16(e);
      m8[r] = mn;
    }
  }
  float zmin = 3.402823466e38f;
  size_t base = (size_t)bh * SP_ + qt * 16;
  #pragma unroll
  for (int r = 0; r < 8; ++r) {
    int srow = qt * 16 + r + half * 8;
    bool valid = srow < S_;
    if ((lane & 15) == 0) {
      mrow[base + r + half * 8] = valid ? m8[r] : 0.f;
      zrow[base + r + half * 8] = valid ? z8[r] : 1.f;
    }
    if (valid) zmin = fminf(zmin, z8[r]);
  }
  zmin = wave_min32(zmin);
  if (lane == 0) atomicMin(&Sc[SL_ZMIN], __float_as_uint(zmin));
}

__global__ void attn_pv_kernel(const i8* __restrict__ q8, const i8* __restrict__ k8,
                               const _Float16* __restrict__ vh,
                               const float* __restrict__ mrow,
                               const float* __restrict__ zrow,
                               unsigned* Sc, float* __restrict__ Of) {
  __shared__ _Float16 pbuf[8][16][32];
  int lane = threadIdx.x & 31, w = threadIdx.x >> 5;
  int t = blockIdx.x * 8 + w;
  int dt = t % 5; int qt = (t / 5) % QT_; int bh = t / (5 * QT_);
  int half = lane >> 4, m = lane & 15;
  const i8* qb = q8 + (size_t)bh * SP_ * KP_;
  const i8* kb = k8 + (size_t)bh * SP_ * KP_;
  float sqk = scale_from(Sc, SL_Q, 127.f) * scale_from(Sc, SL_K, 127.f);
  float s16 = scale_from(Sc, SL_LOG, 32767.f);
  float spq = fmaxf(1.f / __uint_as_float(Sc[SL_ZMIN]), 1e-8f) / 32767.f;
  v8i a0 = load_a_i8(qb, KP_, qt * 16, 0, lane);
  v8i a1 = load_a_i8(qb, KP_, qt * 16, 64, lane);
  float m8[8], rz8[8];
  size_t sbase = (size_t)bh * SP_ + qt * 16;
  #pragma unroll
  for (int r = 0; r < 8; ++r) {
    m8[r]  = mrow[sbase + r + half * 8];
    rz8[r] = 1.f / zrow[sbase + r + half * 8];
  }
  v8f facc = {0.f,0.f,0.f,0.f,0.f,0.f,0.f,0.f};
  for (int kp = 0; kp < QT_ / 2; ++kp) {
    #pragma unroll
    for (int j = 0; j < 2; ++j) {
      int kt = 2 * kp + j;
      v8i b0 = load_b_i8(kb, KP_, kt * 16, 0, lane);
      v8i b1 = load_b_i8(kb, KP_, kt * 16, 64, lane);
      v8i acc = {0,0,0,0,0,0,0,0};
      acc = __builtin_amdgcn_wmma_i32_16x16x64_iu8(true, a0, true, b0, acc, false, false);
      acc = __builtin_amdgcn_wmma_i32_16x16x64_iu8(true, a1, true, b1, acc, false, false);
      int col = kt * 16 + m;
      #pragma unroll
      for (int r = 0; r < 8; ++r) {
        float lq = rintf(((float)acc[r] * sqk) / s16);
        lq = fminf(fmaxf(lq, -32768.f), 32767.f);
        float l = lq * s16 * ATT_SCALE;
        float p = (col < S_) ? __expf(l - m8[r]) * rz8[r] : 0.f;
        float pq = fminf(rintf(p / spq), 32767.f);
        pbuf[w][r + half * 8][j * 16 + m] = (_Float16)(pq * spq);
      }
    }
    __syncthreads();
    v16h pa;
    #pragma unroll
    for (int q = 0; q < 8; ++q) {
      pa[q]     = pbuf[w][m][half * 16 + q];
      pa[8 + q] = pbuf[w][m][half * 16 + 8 + q];
    }
    const _Float16* vb0 = vh + ((size_t)bh * SP_ + kp * 32 + half * 16) * DP_ + dt * 16 + m;
    v16h vb;
    #pragma unroll
    for (int kk = 0; kk < 16; ++kk) vb[kk] = vb0[(size_t)kk * DP_];
    facc = __builtin_amdgcn_wmma_f32_16x16x32_f16(false, pa, false, vb,
                                                  (short)0, facc, false, false);
    __syncthreads();
  }
  float mx = 0.f;
  #pragma unroll
  for (int r = 0; r < 8; ++r) {
    int srow = qt * 16 + r + half * 8;
    int d = dt * 16 + m;
    if (srow < S_ && d < HD_) {
      float v = facc[r];
      Of[((size_t)(bh >> 4) * S_ + srow) * H_ + (bh & 15) * HD_ + d] = v;
      mx = fmaxf(mx, fabsf(v));
    }
  }
  mx = wave_max32(mx);
  if (lane == 0) atomicMax(&Sc[SL_O], __float_as_uint(mx));
}

extern "C" void kernel_launch(void* const* d_in, const int* in_sizes, int n_in,
                              void* d_out, int out_size, void* d_ws, size_t ws_size,
                              hipStream_t stream) {
  (void)in_sizes; (void)n_in; (void)out_size; (void)ws_size;
  const float* hx = (const float*)d_in[0];
  const float* Wq = (const float*)d_in[1]; const float* bq = (const float*)d_in[2];
  const float* Wk = (const float*)d_in[3]; const float* bk = (const float*)d_in[4];
  const float* Wv = (const float*)d_in[5]; const float* bv = (const float*)d_in[6];
  const float* Wo = (const float*)d_in[7]; const float* bo = (const float*)d_in[8];
  float* out = (float*)d_out;

  uint8_t* wsb = (uint8_t*)d_ws; size_t off = 0;
  auto alloc = [&](size_t bytes) -> void* {
    void* p = wsb + off; off = (off + bytes + 255) & ~(size_t)255; return p;
  };
  unsigned* Sc = (unsigned*)alloc(64);
  i8* qx  = (i8*)alloc((size_t)BS_ * H_);
  i8* qWq = (i8*)alloc((size_t)H_ * H_);
  i8* qWk = (i8*)alloc((size_t)H_ * H_);
  i8* qWv = (i8*)alloc((size_t)H_ * H_);
  i8* qWo = (i8*)alloc((size_t)H_ * H_);
  float* Qf = (float*)alloc((size_t)BS_ * H_ * 4);
  float* Kf = (float*)alloc((size_t)BS_ * H_ * 4);
  i8* q8h = (i8*)alloc((size_t)B_ * NH_ * SP_ * KP_);
  i8* k8h = (i8*)alloc((size_t)B_ * NH_ * SP_ * KP_);
  _Float16* vh = (_Float16*)alloc((size_t)B_ * NH_ * SP_ * DP_ * 2);
  float* mrow = (float*)alloc((size_t)B_ * NH_ * SP_ * 4);
  float* zrow = (float*)alloc((size_t)B_ * NH_ * SP_ * 4);
  float* Of = (float*)alloc((size_t)BS_ * H_ * 4);
  i8* qO = (i8*)alloc((size_t)BS_ * H_);

  const size_t nX = (size_t)BS_ * H_, nW = (size_t)H_ * H_;

  init_scales_kernel<<<1, 32, 0, stream>>>(Sc);
  absmax_kernel<<<1280, 256, 0, stream>>>(hx, nX, Sc, SL_X);
  absmax_kernel<<<1280, 256, 0, stream>>>(Wq, nW, Sc, SL_WQ);
  absmax_kernel<<<1280, 256, 0, stream>>>(Wk, nW, Sc, SL_WK);
  absmax_kernel<<<1280, 256, 0, stream>>>(Wv, nW, Sc, SL_WV);
  absmax_kernel<<<1280, 256, 0, stream>>>(Wo, nW, Sc, SL_WO);

  quant8_kernel<<<(int)((nX + 255) / 256), 256, 0, stream>>>(hx, nX, Sc, SL_X, qx);
  quant8_kernel<<<(int)((nW + 255) / 256), 256, 0, stream>>>(Wq, nW, Sc, SL_WQ, qWq);
  quant8_kernel<<<(int)((nW + 255) / 256), 256, 0, stream>>>(Wk, nW, Sc, SL_WK, qWk);
  quant8_kernel<<<(int)((nW + 255) / 256), 256, 0, stream>>>(Wv, nW, Sc, SL_WV, qWv);
  quant8_kernel<<<(int)((nW + 255) / 256), 256, 0, stream>>>(Wo, nW, Sc, SL_WO, qWo);

  hipMemsetAsync(vh, 0, (size_t)B_ * NH_ * SP_ * DP_ * 2, stream);

  const int gemmBlocks = (MTILE_ * NGRP_ + 7) / 8;
  gemm_i8_kernel<<<gemmBlocks, 256, 0, stream>>>(qx, qWq, bq, Sc, SL_X, SL_WQ,
                                                 H_, Qf, nullptr, 0);
  gemm_i8_kernel<<<gemmBlocks, 256, 0, stream>>>(qx, qWk, bk, Sc, SL_X, SL_WK,
                                                 H_, Kf, nullptr, 0);
  gemm_i8_kernel<<<gemmBlocks, 256, 0, stream>>>(qx, qWv, bv, Sc, SL_X, SL_WV,
                                                 H_, nullptr, vh, 1);

  absmax_kernel<<<1280, 256, 0, stream>>>(Qf, nX, Sc, SL_Q);
  absmax_kernel<<<1280, 256, 0, stream>>>(Kf, nX, Sc, SL_K);

  const size_t nQP = (size_t)B_ * NH_ * SP_ * KP_;
  quant_head_kernel<<<(int)(nQP / 256), 256, 0, stream>>>(Qf, Sc, SL_Q, q8h);
  quant_head_kernel<<<(int)(nQP / 256), 256, 0, stream>>>(Kf, Sc, SL_K, k8h);

  attn_logit_amax_kernel<<<B_ * NH_ * QT_ / 8, 256, 0, stream>>>(q8h, k8h, Sc);
  attn_rowstats_kernel<<<B_ * NH_ * QT_ / 8, 256, 0, stream>>>(q8h, k8h, Sc, mrow, zrow);
  attn_pv_kernel<<<B_ * NH_ * QT_ * 5 / 8, 256, 0, stream>>>(q8h, k8h, vh, mrow, zrow, Sc, Of);

  quant8_kernel<<<(int)((nX + 255) / 256), 256, 0, stream>>>(Of, nX, Sc, SL_O, qO);
  gemm_i8_kernel<<<gemmBlocks, 256, 0, stream>>>(qO, qWo, bo, Sc, SL_O, SL_WO,
                                                 H_, out, nullptr, 0);
}
